// MAGE_27049704030297
// MI455X (gfx1250) — compile-verified, matrix operand order checked by reading, not run on hardware
//
#include <hip/hip_runtime.h>
#include <math.h>

typedef __attribute__((ext_vector_type(16))) _Float16 v16h;
typedef __attribute__((ext_vector_type(8)))  float    v8f;

#define NNODES 50000
#define INDIM  128
#define ERAW   800000
#define NEDGE  (ERAW + NNODES)   // reference appends self loops
#define NGRAPH 256
#define NHEAD  2
#define POOLD  448               // 64 + 128 + 256
#define BN_EPS 1e-5f
#define NEG_ATT 0.2f
#define NEG_ACT 0.01f

#define A_STRIDE 36              // 16x32 A tile, padded (36*4B = 144B, 16B aligned, no bank conflicts)
#define B_STRIDE 68              // 32x64 B tile, padded (68*4B = 272B, 16B aligned)

__device__ __forceinline__ float lrelu(float v, float s) { return v > 0.f ? v : v * s; }

// order-preserving float <-> uint key for atomicMax-based segment max
__device__ __forceinline__ unsigned f2key(float f) {
    unsigned u = __float_as_uint(f);
    return (u & 0x80000000u) ? ~u : (u | 0x80000000u);
}
__device__ __forceinline__ float key2f(unsigned k) {
    unsigned u = (k & 0x80000000u) ? (k & 0x7fffffffu) : ~k;
    return __uint_as_float(u);
}

// async global -> LDS copy, 16B per lane (GLOBAL_LOAD_ASYNC_TO_LDS_B128, ASYNCcnt)
__device__ __forceinline__ void async_g2l_b128(unsigned lds_off, const float* g) {
    asm volatile("global_load_async_to_lds_b128 %0, %1, off"
                 :: "v"(lds_off), "v"((unsigned long long)(size_t)g)
                 : "memory");
}
__device__ __forceinline__ void wait_async0() {
    asm volatile("s_wait_asynccnt 0" ::: "memory");
}

__global__ void fill_kernel(float* p, float v, int n) {
    int i = blockIdx.x * blockDim.x + threadIdx.x;
    if (i < n) p[i] = v;
}

// ---------------------------------------------------------------------------
// WMMA GEMM: C[n,m] = A[n,k] @ B[k,m]; fp32 in/out, f16 multiplicands,
// f32 accumulate via v_wmma_f32_16x16x32_f16.
// Block = 128 threads (4 waves); block tile = 16 rows x 64 cols; per k-step the
// block async-copies A(16x32) and B(32x64) tiles to LDS (coalesced 16B lanes),
// waits ASYNCcnt, barriers, then each wave builds its fragments from LDS.
// Requires n%16==0, k%32==0, m%64==0. EXEC all-ones by construction.
// ---------------------------------------------------------------------------
__global__ void wmma_gemm_kernel(const float* __restrict__ A,
                                 const float* __restrict__ B,
                                 float* __restrict__ C, int K, int M) {
    __shared__ float Abuf[16 * A_STRIDE];
    __shared__ float Bbuf[32 * B_STRIDE];

    const int t    = threadIdx.x;
    const int lane = t & 31;
    const int wave = t >> 5;
    const int half = lane >> 4;        // 0: lanes 0-15, 1: lanes 16-31
    const int lr   = lane & 15;        // A row / B col / C col within tile
    const int row0 = blockIdx.x * 16;
    const int colb = blockIdx.y * 64;  // block col base; wave owns colb+wave*16
    const int kb   = half * 8;         // per-half K base (ISA 16-bit layout)

    // cooperative copy coordinates
    const int ar = t >> 3;             // A tile row   (0..15)
    const int ak = (t & 7) * 4;        // A tile k off (0,4,..,28)
    const int bk = t >> 4;             // B tile k base(0..7), +8 per iteration
    const int bc = (t & 15) * 4;       // B tile col   (0,4,..,60)

    v8f acc = {};
    for (int k0 = 0; k0 < K; k0 += 32) {
        // stage tiles into LDS via async copies (16B per lane, fully coalesced)
        async_g2l_b128((unsigned)(size_t)&Abuf[ar * A_STRIDE + ak],
                       A + (size_t)(row0 + ar) * K + k0 + ak);
#pragma unroll
        for (int it = 0; it < 4; ++it) {
            int kk = bk + it * 8;
            async_g2l_b128((unsigned)(size_t)&Bbuf[kk * B_STRIDE + bc],
                           B + (size_t)(k0 + kk) * M + colb + bc);
        }
        wait_async0();
        __syncthreads();

        // fragment assembly per ISA 16-bit WMMA layouts
        v16h a, b;
#pragma unroll
        for (int i = 0; i < 8; ++i) {
            // VGPR i holds K pair: i<4 -> kb+2i, else kb+16+2(i-4)
            int kk = kb + ((i < 4) ? (2 * i) : (16 + 2 * (i - 4)));
            a[2 * i]     = (_Float16)Abuf[lr * A_STRIDE + kk];
            a[2 * i + 1] = (_Float16)Abuf[lr * A_STRIDE + kk + 1];
            b[2 * i]     = (_Float16)Bbuf[kk * B_STRIDE + wave * 16 + lr];
            b[2 * i + 1] = (_Float16)Bbuf[(kk + 1) * B_STRIDE + wave * 16 + lr];
        }
        acc = __builtin_amdgcn_wmma_f32_16x16x32_f16(
            /*neg_a=*/false, a, /*neg_b=*/false, b,
            /*c_mod=*/(short)0, acc, /*reuse_a=*/false, /*reuse_b=*/false);
        __syncthreads();
    }
#pragma unroll
    for (int r = 0; r < 8; ++r)       // C/D layout: VGPR r -> M=r (+8 for hi half)
        C[(size_t)(row0 + r + half * 8) * M + colb + wave * 16 + lr] = acc[r];
}

// ---------------------------------------------------------------------------
// per-(node,head) attention logits: al_s = <h, a_src>, al_d = <h, a_dst>
// ---------------------------------------------------------------------------
__global__ void att_kernel(const float* __restrict__ h,
                           const float* __restrict__ a_src,
                           const float* __restrict__ a_dst,
                           float* al_s, float* al_d, int C) {
    int i = blockIdx.x * blockDim.x + threadIdx.x;
    if (i >= NNODES * NHEAD) return;
    int n = i >> 1, hh = i & 1;
    const float* hp = h + (size_t)n * (NHEAD * C) + hh * C;
    const float* as = a_src + hh * C;
    const float* ad = a_dst + hh * C;
    float s = 0.f, d = 0.f;
    for (int c = 0; c < C; ++c) { float v = hp[c]; s += v * as[c]; d += v * ad[c]; }
    al_s[i] = s;
    al_d[i] = d;
}

__device__ __forceinline__ void edge_ends(const int* __restrict__ ei, int e,
                                          int& s, int& d) {
    if (e < ERAW) { s = ei[e]; d = ei[ERAW + e]; }
    else          { s = d = e - ERAW; }          // self loop
}

// pass 1: per-dst segment max of leaky_relu(al_s[src]+al_d[dst], 0.2)
__global__ void edge_max_kernel(const int* __restrict__ ei,
                                const float* __restrict__ al_s,
                                const float* __restrict__ al_d,
                                unsigned* mkey) {
    int i = blockIdx.x * blockDim.x + threadIdx.x;
    if (i >= NEDGE * NHEAD) return;
    int e = i >> 1, hh = i & 1, s, d;
    edge_ends(ei, e, s, d);
    float v = lrelu(al_s[s * NHEAD + hh] + al_d[d * NHEAD + hh], NEG_ATT);
    atomicMax(&mkey[d * NHEAD + hh], f2key(v));
}

// pass 2: ee = exp(e - max[dst]); segment-sum into ssum
__global__ void edge_exp_kernel(const int* __restrict__ ei,
                                const float* __restrict__ al_s,
                                const float* __restrict__ al_d,
                                const unsigned* __restrict__ mkey,
                                float* eebuf, float* ssum) {
    int i = blockIdx.x * blockDim.x + threadIdx.x;
    if (i >= NEDGE * NHEAD) return;
    int e = i >> 1, hh = i & 1, s, d;
    edge_ends(ei, e, s, d);
    float v = lrelu(al_s[s * NHEAD + hh] + al_d[d * NHEAD + hh], NEG_ATT);
    float m = key2f(mkey[d * NHEAD + hh]);
    float ee = expf(v - m);
    eebuf[i] = ee;
    atomicAdd(&ssum[d * NHEAD + hh], ee);
}

// pass 3: out[dst] += h[src] * ee/(sum[dst]+1e-16)  (the HBM-dominant scatter)
__global__ void edge_agg_kernel(const int* __restrict__ ei,
                                const float* __restrict__ h,
                                const float* __restrict__ eebuf,
                                const float* __restrict__ ssum,
                                float* out, int M, int C) {
    int total = NEDGE * M;
    for (int idx = blockIdx.x * blockDim.x + threadIdx.x; idx < total;
         idx += gridDim.x * blockDim.x) {
        int e = idx / M, f = idx - e * M, s, d;
        edge_ends(ei, e, s, d);
        int hh = f / C;
        float alpha = eebuf[e * NHEAD + hh] / (ssum[d * NHEAD + hh] + 1e-16f);
        atomicAdd(&out[(size_t)d * M + f], h[(size_t)s * M + f] * alpha);
    }
}

// BN stats: one workgroup per feature column, LDS tree reduce
__global__ void bn_stats_kernel(const float* __restrict__ x,
                                const float* __restrict__ bias,
                                float* meanb, float* varb, int M) {
    int j = blockIdx.x;
    float s = 0.f, s2 = 0.f;
    float bj = bias[j];
    for (int i = threadIdx.x; i < NNODES; i += blockDim.x) {
        float v = x[(size_t)i * M + j] + bj;
        s += v; s2 += v * v;
    }
    __shared__ float sh[256], sh2[256];
    sh[threadIdx.x] = s; sh2[threadIdx.x] = s2;
    __syncthreads();
    for (int off = 128; off > 0; off >>= 1) {
        if (threadIdx.x < off) {
            sh[threadIdx.x]  += sh[threadIdx.x + off];
            sh2[threadIdx.x] += sh2[threadIdx.x + off];
        }
        __syncthreads();
    }
    if (threadIdx.x == 0) {
        float mu = sh[0] / (float)NNODES;
        meanb[j] = mu;
        varb[j]  = sh2[0] / (float)NNODES - mu * mu;   // biased var, as jnp.var
    }
}

__global__ void bn_apply_kernel(const float* __restrict__ x,
                                const float* __restrict__ bias,
                                const float* __restrict__ meanb,
                                const float* __restrict__ varb,
                                const float* __restrict__ g,
                                const float* __restrict__ be,
                                float* out, int M) {
    int idx = blockIdx.x * blockDim.x + threadIdx.x;
    if (idx >= NNODES * M) return;
    int j = idx % M;
    float v = (x[idx] + bias[j] - meanb[j]) * rsqrtf(varb[j] + BN_EPS) * g[j] + be[j];
    out[idx] = lrelu(v, NEG_ACT);
}

__global__ void pool_kernel(const float* __restrict__ x,
                            const int* __restrict__ batch,
                            float* pooled, int M, int off) {
    int total = NNODES * M;
    for (int idx = blockIdx.x * blockDim.x + threadIdx.x; idx < total;
         idx += gridDim.x * blockDim.x) {
        int n = idx / M, f = idx - n * M;
        atomicAdd(&pooled[(size_t)batch[n] * POOLD + off + f], x[idx]);
    }
}

__global__ void dense_kernel(const float* __restrict__ in,
                             const float* __restrict__ W,
                             const float* __restrict__ b,
                             float* out, int n, int k, int m, int act) {
    int idx = blockIdx.x * blockDim.x + threadIdx.x;
    if (idx >= n * m) return;
    int i = idx / m, j = idx - i * m;
    float v = b[j];
    for (int kk = 0; kk < k; ++kk) v += in[(size_t)i * k + kk] * W[(size_t)kk * m + j];
    out[idx] = act ? lrelu(v, NEG_ACT) : v;
}

// ---------------------------------------------------------------------------
static inline int cdiv(int a, int b) { return (a + b - 1) / b; }

static void run_gat_layer(hipStream_t stream, const int* ei,
                          const float* xin, int K, int M, int C,
                          const float* W, const float* a_src, const float* a_dst,
                          const float* bias, const float* g, const float* be,
                          float* h_buf, float* agg_buf, float* al_s, float* al_d,
                          unsigned* mkey, float* ssum, float* eebuf,
                          float* meanb, float* varb, float* xout) {
    // h = x @ W (WMMA path, async-LDS staged)
    wmma_gemm_kernel<<<dim3(NNODES / 16, M / 64), 128, 0, stream>>>(xin, W, h_buf, K, M);
    // attention logits
    att_kernel<<<cdiv(NNODES * NHEAD, 256), 256, 0, stream>>>(h_buf, a_src, a_dst, al_s, al_d, C);
    // zero accumulators (poison-safe, graph-capture-safe)
    fill_kernel<<<cdiv(NNODES * NHEAD, 256), 256, 0, stream>>>((float*)mkey, 0.f, NNODES * NHEAD);
    fill_kernel<<<cdiv(NNODES * NHEAD, 256), 256, 0, stream>>>(ssum, 0.f, NNODES * NHEAD);
    fill_kernel<<<cdiv(NNODES * M, 256), 256, 0, stream>>>(agg_buf, 0.f, NNODES * M);
    // softmax over dst (3 edge passes)
    edge_max_kernel<<<cdiv(NEDGE * NHEAD, 256), 256, 0, stream>>>(ei, al_s, al_d, mkey);
    edge_exp_kernel<<<cdiv(NEDGE * NHEAD, 256), 256, 0, stream>>>(ei, al_s, al_d, mkey, eebuf, ssum);
    edge_agg_kernel<<<8192, 256, 0, stream>>>(ei, h_buf, eebuf, ssum, agg_buf, M, C);
    // bias + batchnorm + leaky_relu(0.01)
    bn_stats_kernel<<<M, 256, 0, stream>>>(agg_buf, bias, meanb, varb, M);
    bn_apply_kernel<<<cdiv(NNODES * M, 256), 256, 0, stream>>>(agg_buf, bias, meanb, varb, g, be, xout, M);
}

extern "C" void kernel_launch(void* const* d_in, const int* in_sizes, int n_in,
                              void* d_out, int out_size, void* d_ws, size_t ws_size,
                              hipStream_t stream) {
    const float* x   = (const float*)d_in[0];
    const int*   ei  = (const int*)d_in[1];
    // d_in[2] = edge_weigth: unused by the reference
    const int*   bat = (const int*)d_in[3];
    const float *W1 = (const float*)d_in[4],  *as1 = (const float*)d_in[5],  *ad1 = (const float*)d_in[6];
    const float *b1 = (const float*)d_in[7],  *g1  = (const float*)d_in[8],  *be1 = (const float*)d_in[9];
    const float *W2 = (const float*)d_in[10], *as2 = (const float*)d_in[11], *ad2 = (const float*)d_in[12];
    const float *b2 = (const float*)d_in[13], *g2  = (const float*)d_in[14], *be2 = (const float*)d_in[15];
    const float *W3 = (const float*)d_in[16], *as3 = (const float*)d_in[17], *ad3 = (const float*)d_in[18];
    const float *b3 = (const float*)d_in[19], *g3  = (const float*)d_in[20], *be3 = (const float*)d_in[21];
    const float *Wc1 = (const float*)d_in[22], *bc1 = (const float*)d_in[23];
    const float *Wc2 = (const float*)d_in[24], *bc2 = (const float*)d_in[25];
    const float *Wc3 = (const float*)d_in[26], *bc3 = (const float*)d_in[27];
    float* out = (float*)d_out;

    // workspace carve-out
    char* w = (char*)d_ws;
    auto carve = [&](size_t bytes) {
        void* p = (void*)w;
        w += (bytes + 255) & ~(size_t)255;
        return p;
    };
    float*    h_buf   = (float*)carve((size_t)NNODES * 256 * 4);
    float*    agg_buf = (float*)carve((size_t)NNODES * 256 * 4);
    float*    x1      = (float*)carve((size_t)NNODES * 64 * 4);
    float*    x2      = (float*)carve((size_t)NNODES * 128 * 4);
    float*    x3      = (float*)carve((size_t)NNODES * 256 * 4);
    float*    al_s    = (float*)carve((size_t)NNODES * NHEAD * 4);
    float*    al_d    = (float*)carve((size_t)NNODES * NHEAD * 4);
    unsigned* mkey    = (unsigned*)carve((size_t)NNODES * NHEAD * 4);
    float*    ssum    = (float*)carve((size_t)NNODES * NHEAD * 4);
    float*    eebuf   = (float*)carve((size_t)NEDGE * NHEAD * 4);
    float*    meanb   = (float*)carve(256 * 4);
    float*    varb    = (float*)carve(256 * 4);
    float*    pooled  = (float*)carve((size_t)NGRAPH * POOLD * 4);
    float*    m1      = (float*)carve((size_t)NGRAPH * 128 * 4);
    float*    m2      = (float*)carve((size_t)NGRAPH * 64 * 4);

    // three GAT + BN + leaky layers
    run_gat_layer(stream, ei, x,  INDIM, 64,  32,  W1, as1, ad1, b1, g1, be1,
                  h_buf, agg_buf, al_s, al_d, mkey, ssum, eebuf, meanb, varb, x1);
    run_gat_layer(stream, ei, x1, 64,    128, 64,  W2, as2, ad2, b2, g2, be2,
                  h_buf, agg_buf, al_s, al_d, mkey, ssum, eebuf, meanb, varb, x2);
    run_gat_layer(stream, ei, x2, 128,   256, 128, W3, as3, ad3, b3, g3, be3,
                  h_buf, agg_buf, al_s, al_d, mkey, ssum, eebuf, meanb, varb, x3);

    // per-graph sum pooling of [x1 | x2 | x3]
    fill_kernel<<<cdiv(NGRAPH * POOLD, 256), 256, 0, stream>>>(pooled, 0.f, NGRAPH * POOLD);
    pool_kernel<<<4096, 256, 0, stream>>>(x1, bat, pooled, 64, 0);
    pool_kernel<<<4096, 256, 0, stream>>>(x2, bat, pooled, 128, 64);
    pool_kernel<<<4096, 256, 0, stream>>>(x3, bat, pooled, 256, 192);

    // classifier MLP
    dense_kernel<<<cdiv(NGRAPH * 128, 256), 256, 0, stream>>>(pooled, Wc1, bc1, m1, NGRAPH, POOLD, 128, 1);
    dense_kernel<<<cdiv(NGRAPH * 64, 256), 256, 0, stream>>>(m1, Wc2, bc2, m2, NGRAPH, 128, 64, 1);
    dense_kernel<<<cdiv(NGRAPH * 2, 256), 256, 0, stream>>>(m2, Wc3, bc3, out, NGRAPH, 64, 2, 0);
}
